// FineMatching_23794118820198
// MI455X (gfx1250) — matrix operand order repaired and checked by reading, not compiled, not run on hardware
//
#include <hip/hip_runtime.h>
#include <math.h>

typedef __attribute__((ext_vector_type(16))) _Float16 v16h;
typedef __attribute__((ext_vector_type(8)))  float    v8f;

union Frag16 { v16h h; uint4 q[2]; };

#define M_TOT   8192
#define SIM_SCALE (10.0f / 96.0f)   // C^-1 / TEMP
#define K1      1728                // 192 ch * 9 taps
#define LRELU(x) ((x) > 0.0f ? (x) : 0.01f * (x))

// Padded LDS row strides (in halves): +8 halves makes per-lane dword stride
// coprime-enough with 64 banks (36/52 mod 64 -> 16 distinct bank groups).
#define SIM_ROW   104               // 96 + 8
#define XPAD_M    4808              // 25*192 + 8
#define OUT1_M    1736              // 9*192 + 8

#define XPAD_BYTES (16 * XPAD_M * 2)          // 153856
#define OUT1_OFF   XPAD_BYTES
#define OUT1_BYTES (16 * OUT1_M * 2)          // 55552
#define OUT2_OFF   (OUT1_OFF + OUT1_BYTES)    // 209408
#define OUT2_BYTES (16 * 192 * 4)             // 12288
#define META_OFF   (OUT2_OFF + OUT2_BYTES)    // 221696
#define SMEM_TOTAL (META_OFF + 16 * 5 * 4)    // 222016

// ---------------------------------------------------------------------------
// Kernel 0: reorder conv weights f32 -> f16, [o][k] with k = (ky*3+kx)*192 + i
// so a WMMA B-fragment (col o, 16 consecutive k) is two contiguous uint4 loads.
// ---------------------------------------------------------------------------
__global__ __launch_bounds__(256) void prep_weights_kernel(
    const float* __restrict__ w1, const float* __restrict__ w2,
    _Float16* __restrict__ w1f, _Float16* __restrict__ w2f)
{
    int e = blockIdx.x * blockDim.x + threadIdx.x;
    const int n = 192 * K1;
    if (e >= 2 * n) return;
    const float* src = (e < n) ? w1 : w2;
    _Float16*    dst = (e < n) ? w1f : w2f;
    int r = (e < n) ? e : e - n;
    int o = r / K1;
    int k = r - o * K1;
    int q = k / 192;             // tap ky*3+kx
    int i = k - q * 192;         // input channel
    int ky = q / 3, kx = q - 3 * ky;
    dst[r] = (_Float16)src[((o * 192 + i) * 3 + ky) * 3 + kx];
}

// ---------------------------------------------------------------------------
// Kernel 1: per-m correlation (WMMA), dual softmax, conf out, argmax/mconf.
// One block (256 thr = 8 waves) per m. sim = F0(64x96) * F1^T(96x64).
// ---------------------------------------------------------------------------
__global__ __launch_bounds__(256) void sim_softmax_kernel(
    const float* __restrict__ f0, const float* __restrict__ f1,
    float* __restrict__ conf_out, float* __restrict__ mconf_out,
    float* __restrict__ valid_out, int* __restrict__ idx_out)
{
    __shared__ _Float16 A0[64 * SIM_ROW];
    __shared__ _Float16 B1[64 * SIM_ROW];
    __shared__ float simb[64 * 64];
    __shared__ float rmax[64], rsum[64], cmax[64], csum[64];
    __shared__ float redv[256];
    __shared__ int   redi[256];

    const int m    = blockIdx.x;
    const int t    = threadIdx.x;
    const int lane = t & 31;
    const int wave = t >> 5;
    const int ln   = lane & 15;
    const int hi   = lane >> 4;

    const float* p0 = f0 + (size_t)m * 64 * 96;
    const float* p1 = f1 + (size_t)m * 64 * 96;
    for (int e = t; e < 64 * 96; e += 256) {
        int l = e / 96, c = e - l * 96;
        A0[l * SIM_ROW + c] = (_Float16)p0[e];
        B1[l * SIM_ROW + c] = (_Float16)p1[e];
    }
    __syncthreads();

    // 16 output tiles (4x4 of 16x16); 8 waves -> 2 tiles each (same ct, so the
    // compiler keeps B fragments live across both); K = 3 x 32.
    for (int tile = wave; tile < 16; tile += 8) {
        int rt = tile >> 2, ct = tile & 3;
        v8f acc = {};
#pragma unroll
        for (int kc = 0; kc < 3; ++kc) {
            Frag16 a, b;
            int arow = rt * 16 + ln;
            int ab   = kc * 32 + hi * 16;
            a.q[0] = *(const uint4*)&A0[arow * SIM_ROW + ab];
            a.q[1] = *(const uint4*)&A0[arow * SIM_ROW + ab + 8];
            int bcol = ct * 16 + ln;
            int bb   = kc * 32 + hi * 16;
            b.q[0] = *(const uint4*)&B1[bcol * SIM_ROW + bb];
            b.q[1] = *(const uint4*)&B1[bcol * SIM_ROW + bb + 8];
            acc = __builtin_amdgcn_wmma_f32_16x16x32_f16(
                false, a.h, false, b.h, (short)0, acc, false, false);
        }
#pragma unroll
        for (int v = 0; v < 8; ++v) {
            int rr = rt * 16 + v + hi * 8;
            int cc = ct * 16 + ln;
            simb[rr * 64 + cc] = acc[v] * SIM_SCALE;
        }
    }
    __syncthreads();

    // Softmax stats, 4 threads per row / per column (all 256 threads busy).
    const int rid = t >> 2, part = t & 3;
    {   // row max
        float mx = -1e30f;
        int base = rid * 64 + part * 16;
#pragma unroll
        for (int k = 0; k < 16; ++k) mx = fmaxf(mx, simb[base + k]);
        redv[t] = mx;
    }
    __syncthreads();
    if (part == 0)
        rmax[rid] = fmaxf(fmaxf(redv[t], redv[t + 1]), fmaxf(redv[t + 2], redv[t + 3]));
    __syncthreads();
    {   // row expsum
        float mx = rmax[rid], sm = 0.0f;
        int base = rid * 64 + part * 16;
#pragma unroll
        for (int k = 0; k < 16; ++k) sm += __expf(simb[base + k] - mx);
        redv[t] = sm;
    }
    __syncthreads();
    if (part == 0) rsum[rid] = redv[t] + redv[t + 1] + redv[t + 2] + redv[t + 3];
    __syncthreads();
    {   // col max
        float mx = -1e30f;
#pragma unroll
        for (int k = 0; k < 16; ++k) mx = fmaxf(mx, simb[(part * 16 + k) * 64 + rid]);
        redv[t] = mx;
    }
    __syncthreads();
    if (part == 0)
        cmax[rid] = fmaxf(fmaxf(redv[t], redv[t + 1]), fmaxf(redv[t + 2], redv[t + 3]));
    __syncthreads();
    {   // col expsum
        float mx = cmax[rid], sm = 0.0f;
#pragma unroll
        for (int k = 0; k < 16; ++k) sm += __expf(simb[(part * 16 + k) * 64 + rid] - mx);
        redv[t] = sm;
    }
    __syncthreads();
    if (part == 0) csum[rid] = redv[t] + redv[t + 1] + redv[t + 2] + redv[t + 3];
    __syncthreads();

    // conf = softmax_l * softmax_s = exp(2*sim - rmax - cmax)/(rsum*csum)
    float bv = -1.0f; int bi = 0;
    float* cdst = conf_out + (size_t)m * 4096;
    for (int k2 = 0; k2 < 16; ++k2) {
        int e = t + 256 * k2;
        int l = e >> 6, s = e & 63;
        float sv = simb[l * 64 + s];
        float cv = __expf(2.0f * sv - rmax[l] - cmax[s]) / (rsum[l] * csum[s]);
        cdst[e] = cv;
        if (cv > bv) { bv = cv; bi = e; }
    }
    redv[t] = bv; redi[t] = bi;
    __syncthreads();
    for (int s2 = 128; s2 > 0; s2 >>= 1) {
        if (t < s2) {
            float ov = redv[t + s2]; int oi = redi[t + s2];
            if (ov > redv[t] || (ov == redv[t] && oi < redi[t])) {
                redv[t] = ov; redi[t] = oi;
            }
        }
        __syncthreads();
    }
    if (t == 0) {
        float mc = redv[0];
        bool ok  = mc > 0.01f;
        mconf_out[m] = ok ? mc : 0.0f;
        valid_out[m] = ok ? 1.0f : 0.0f;
        idx_out[m]   = redi[0];
    }
}

// ---------------------------------------------------------------------------
// Kernel 2: gather 3x3 patches (zero-padded to 5x5 in LDS), conv1 (WMMA,
// implicit GEMM per output position, 2 o-tiles per A-fragment load), conv2
// (WMMA), conv3 + tanh heads. One block (256 thr) per 16 matches.
// Dynamic LDS ~217 KB (<= 320 KB per WGP).
// ---------------------------------------------------------------------------
__global__ __launch_bounds__(256) void fine_kernel(
    const float* __restrict__ f0w, const float* __restrict__ f1w,
    const int* __restrict__ b_ids, const int* __restrict__ i_idsc,
    const int* __restrict__ j_idsc,
    const _Float16* __restrict__ w1f, const _Float16* __restrict__ w2f,
    const float* __restrict__ w3,
    const int* __restrict__ idx_in,
    float* __restrict__ mk0_out, float* __restrict__ mk1_out)
{
    extern __shared__ char smem[];
    _Float16* xpad = (_Float16*)smem;               // [16 m][25 pos][192 ch] (+pad)
    _Float16* out1 = (_Float16*)(smem + OUT1_OFF);  // [16 m][9 p][192 o] (+pad)
    float*    out2 = (float*)(smem + OUT2_OFF);     // [16 m][192]
    int*      meta = (int*)(smem + META_OFF);       // [16 m][5]

    const int m0   = blockIdx.x * 16;
    const int t    = threadIdx.x;
    const int lane = t & 31;
    const int wave = t >> 5;
    const int ln   = lane & 15;
    const int hi   = lane >> 4;

    // Per-m metadata: b, mk0x, mk0y, mk1x, mk1y (unscaled fine coords).
    if (t < 16) {
        int m  = m0 + t;
        int id = idx_in[m];
        int ii = id >> 6, jj = id & 63;
        int ic = i_idsc[m], jc = j_idsc[m];
        meta[t * 5 + 0] = b_ids[m];
        meta[t * 5 + 1] = (ic % 80) * 4 + (ii & 7) - 4;
        meta[t * 5 + 2] = (ic / 80) * 4 + (ii >> 3) - 4;
        meta[t * 5 + 3] = (jc % 80) * 4 + (jj & 7) - 4;
        meta[t * 5 + 4] = (jc / 80) * 4 + (jj >> 3) - 4;
    }
    // Zero the (padded) patch buffer.
    uint32_t* xz = (uint32_t*)xpad;
    for (int e = t; e < XPAD_BYTES / 4; e += 256) xz[e] = 0u;
    __syncthreads();

    // Gather interior 3x3 patches (192 ch = f0 96 + f1 96), convert to f16.
    for (int e = t; e < 16 * 9 * 192; e += 256) {
        int ml  = e / K1;
        int r2  = e - ml * K1;
        int pos = r2 / 192;
        int ch  = r2 - pos * 192;
        int r = pos / 3, c = pos - 3 * r;
        int b = meta[ml * 5 + 0];
        int gx, gy;
        const float* base;
        int chl;
        if (ch < 96) {
            gx = meta[ml * 5 + 1] + c - 1; gy = meta[ml * 5 + 2] + r - 1;
            base = f0w; chl = ch;
        } else {
            gx = meta[ml * 5 + 3] + c - 1; gy = meta[ml * 5 + 4] + r - 1;
            base = f1w; chl = ch - 96;
        }
        float v = 0.0f;
        if (gx >= 0 && gy >= 0 && gx < 320 && gy < 320)
            v = base[(((size_t)b * 96 + chl) * 320 + gy) * 320 + gx];
        xpad[ml * XPAD_M + ((r + 1) * 5 + (c + 1)) * 192 + ch] = (_Float16)v;
    }
    __syncthreads();

    // conv1: 9 positions x 6 o-tile-pairs = 54 jobs; each job keeps two
    // accumulators so every A-fragment (LDS) feeds two WMMAs. K = 54 x 32.
    for (int job = wave; job < 54; job += 8) {
        int p   = job / 6;
        int op  = job - 6 * p;
        int ot0 = op * 2, ot1 = ot0 + 1;
        int py = p / 3, px = p - 3 * py;
        const _Float16* wb0 = w1f + (size_t)(ot0 * 16 + ln) * K1 + hi * 16;
        const _Float16* wb1 = w1f + (size_t)(ot1 * 16 + ln) * K1 + hi * 16;
        __builtin_prefetch(wb0, 0, 0);
        __builtin_prefetch(wb1, 0, 0);
        v8f acc0 = {}, acc1 = {};
        for (int kc = 0; kc < 54; ++kc) {
            int q     = kc / 6;
            int ibase = (kc - 6 * q) * 32;
            int ky = q / 3, kx = q - 3 * ky;
            int yy = py + ky, xx = px + kx;
            Frag16 a, b0, b1;
            int abase = ln * XPAD_M + (yy * 5 + xx) * 192 + ibase + hi * 16;
            a.q[0] = *(const uint4*)&xpad[abase];
            a.q[1] = *(const uint4*)&xpad[abase + 8];
            const _Float16* w0 = wb0 + kc * 32;
            const _Float16* w1p = wb1 + kc * 32;
            b0.q[0] = *(const uint4*)w0;  b0.q[1] = *(const uint4*)(w0 + 8);
            b1.q[0] = *(const uint4*)w1p; b1.q[1] = *(const uint4*)(w1p + 8);
            acc0 = __builtin_amdgcn_wmma_f32_16x16x32_f16(
                false, a.h, false, b0.h, (short)0, acc0, false, false);
            acc1 = __builtin_amdgcn_wmma_f32_16x16x32_f16(
                false, a.h, false, b1.h, (short)0, acc1, false, false);
        }
#pragma unroll
        for (int v = 0; v < 8; ++v) {
            int mrow = v + hi * 8;
            float x0 = LRELU(acc0[v]);
            float x1 = LRELU(acc1[v]);
            out1[mrow * OUT1_M + p * 192 + ot0 * 16 + ln] = (_Float16)x0;
            out1[mrow * OUT1_M + p * 192 + ot1 * 16 + ln] = (_Float16)x1;
        }
    }
    __syncthreads();

    // conv2 (stride3 VALID == full reduction over p,j): 6 o-tile-pair jobs.
    for (int job = wave; job < 6; job += 8) {
        int ot0 = job * 2, ot1 = ot0 + 1;
        const _Float16* wb0 = w2f + (size_t)(ot0 * 16 + ln) * K1 + hi * 16;
        const _Float16* wb1 = w2f + (size_t)(ot1 * 16 + ln) * K1 + hi * 16;
        __builtin_prefetch(wb0, 0, 0);
        __builtin_prefetch(wb1, 0, 0);
        v8f acc0 = {}, acc1 = {};
        for (int kc = 0; kc < 54; ++kc) {
            int p2    = kc / 6;
            int jbase = (kc - 6 * p2) * 32;
            Frag16 a, b0, b1;
            int abase = ln * OUT1_M + p2 * 192 + jbase + hi * 16;
            a.q[0] = *(const uint4*)&out1[abase];
            a.q[1] = *(const uint4*)&out1[abase + 8];
            const _Float16* w0 = wb0 + kc * 32;
            const _Float16* w1p = wb1 + kc * 32;
            b0.q[0] = *(const uint4*)w0;  b0.q[1] = *(const uint4*)(w0 + 8);
            b1.q[0] = *(const uint4*)w1p; b1.q[1] = *(const uint4*)(w1p + 8);
            acc0 = __builtin_amdgcn_wmma_f32_16x16x32_f16(
                false, a.h, false, b0.h, (short)0, acc0, false, false);
            acc1 = __builtin_amdgcn_wmma_f32_16x16x32_f16(
                false, a.h, false, b1.h, (short)0, acc1, false, false);
        }
#pragma unroll
        for (int v = 0; v < 8; ++v) {
            int mrow = v + hi * 8;
            out2[mrow * 192 + ot0 * 16 + ln] = LRELU(acc0[v]);
            out2[mrow * 192 + ot1 * 16 + ln] = LRELU(acc1[v]);
        }
    }
    __syncthreads();

    // conv3 (4x192 1x1) + tanh heads + final keypoint coordinates.
    if (t < 64) {
        int ml  = t >> 2;
        int jj2 = t & 3;
        const float* wrow = w3 + jj2 * 192;
        float acc = 0.0f;
        for (int o = 0; o < 192; ++o) acc += out2[ml * 192 + o] * wrow[o];
        float sub = tanhf(acc) * 0.5f;
        int m = m0 + ml;
        float base;
        float* dst;
        if (jj2 == 0)      { base = (float)meta[ml * 5 + 1]; dst = mk0_out + m * 2 + 0; }
        else if (jj2 == 1) { base = (float)meta[ml * 5 + 2]; dst = mk0_out + m * 2 + 1; }
        else if (jj2 == 2) { base = (float)meta[ml * 5 + 3]; dst = mk1_out + m * 2 + 0; }
        else               { base = (float)meta[ml * 5 + 4]; dst = mk1_out + m * 2 + 1; }
        *dst = (base + sub) * 2.0f;   // scale = HW0_I/HW0_F = 640/320
    }
}

// ---------------------------------------------------------------------------
extern "C" void kernel_launch(void* const* d_in, const int* in_sizes, int n_in,
                              void* d_out, int out_size, void* d_ws, size_t ws_size,
                              hipStream_t stream)
{
    const float* f0u  = (const float*)d_in[0];
    const float* f1u  = (const float*)d_in[1];
    const float* f0w  = (const float*)d_in[2];
    const float* f1w  = (const float*)d_in[3];
    const int*   bids = (const int*)d_in[4];
    const int*   iidc = (const int*)d_in[5];
    const int*   jidc = (const int*)d_in[6];
    const float* w1   = (const float*)d_in[7];
    const float* w2   = (const float*)d_in[8];
    const float* w3   = (const float*)d_in[9];
    float* out = (float*)d_out;

    // Workspace layout: w1f16 (648 KB) | w2f16 (648 KB) | idx (32 KB)
    char* ws = (char*)d_ws;
    _Float16* w1f    = (_Float16*)ws;
    _Float16* w2f    = (_Float16*)(ws + 663552);
    int*      idxbuf = (int*)(ws + 1327104);

    const int M = M_TOT;
    float* conf  = out;
    float* mconf = out + (size_t)M * 4096;
    float* mk0   = mconf + M;
    float* mk1   = mk0 + 2 * M;
    float* valid = mk1 + 2 * M;

    prep_weights_kernel<<<(2 * 192 * K1 + 255) / 256, 256, 0, stream>>>(w1, w2, w1f, w2f);
    sim_softmax_kernel<<<M, 256, 0, stream>>>(f0u, f1u, conf, mconf, valid, idxbuf);
    fine_kernel<<<M / 16, 256, SMEM_TOTAL, stream>>>(f0w, f1w, bids, iidc, jidc,
                                                     w1f, w2f, w3, idxbuf, mk0, mk1);
}
